// PoolBondFeatures_57956288692318
// MI455X (gfx1250) — compile-verified, matrix operand order checked by reading, not run on hardware
//
#include <hip/hip_runtime.h>
#include <hip/hip_bf16.h>
#include <stdint.h>

// ---------------------------------------------------------------------------
// Types matching gfx1250 WMMA operand register images
// ---------------------------------------------------------------------------
typedef __bf16 bf16_t;
typedef __attribute__((ext_vector_type(16))) __bf16 v16bf;  // A/B fragment (16x32 / 32x16 bf16)
typedef __attribute__((ext_vector_type(8)))  float  v8f;    // C/D fragment (16x16 f32)
typedef __attribute__((ext_vector_type(4)))  int    v4i;    // 16-byte load unit

#define D_DIM    128
#define KTILES   4              // 128 / 32 (K per WMMA)
#define NTILES   8              // 128 / 16 (N per WMMA)
#define FRAG_B   1024           // one B fragment: 32 lanes * 32 bytes
#define MATFRAGS (KTILES * NTILES)          // 32 fragments per 128x128 matrix
#define WFRAG_B  (3 * MATFRAGS * FRAG_B)    // W1a, W1b, W2 -> 96 KB
#define SROW_B   (D_DIM * 2)                // one bf16 row of the s buffer
#define SBUF_B   (16 * SROW_B)              // 4 KB per wave
#define NWAVES   8

union Frag { v16bf m; v4i q[2]; };

__device__ __forceinline__ bf16_t f32_to_bf16_rne(float x) {
  uint32_t u = __builtin_bit_cast(uint32_t, x);
  uint32_t r = u + 0x7FFFu + ((u >> 16) & 1u);   // round-to-nearest-even
  return __builtin_bit_cast(bf16_t, (uint16_t)(r >> 16));
}

// ---------------------------------------------------------------------------
// Prep kernel 1: h (N x 128 f32) -> bf16 (stays resident in 192MB L2: 12.8MB)
// ---------------------------------------------------------------------------
__global__ void convert_h_bf16(const float* __restrict__ h,
                               bf16_t* __restrict__ hbf, int n) {
  int i = (blockIdx.x * 256 + threadIdx.x) * 4;
  if (i + 3 < n) {
    float4 v = *(const float4*)(h + i);
    hbf[i]     = f32_to_bf16_rne(v.x);
    hbf[i + 1] = f32_to_bf16_rne(v.y);
    hbf[i + 2] = f32_to_bf16_rne(v.z);
    hbf[i + 3] = f32_to_bf16_rne(v.w);
  } else {
    for (; i < n; ++i) hbf[i] = f32_to_bf16_rne(h[i]);
  }
}

// ---------------------------------------------------------------------------
// Prep kernel 2: pack W1a (W1 rows 0..127), W1b (rows 128..255), W2 into the
// exact per-lane B-fragment register image:
//   frag element (lane, halfword j): K_local = (v<4 ? 2v : 16+2(v-4)) + pair
//                                    + (lane>=16 ? 8 : 0),  N = 16*nt + lane&15
// Layout: frags[((mat*32 + kt*8 + nt)*32 + lane)*16 + j]
// ---------------------------------------------------------------------------
__global__ void pack_weight_frags(const float* __restrict__ W1,
                                  const float* __restrict__ W2,
                                  bf16_t* __restrict__ frags) {
  int id = blockIdx.x * 256 + threadIdx.x;
  if (id >= 3 * MATFRAGS * 32 * 16) return;
  int j    = id & 15;
  int lane = (id >> 4) & 31;
  int f    = (id >> 9) & 31;
  int mat  = id >> 14;
  int kt = f >> 3, nt = f & 7;
  int v = j >> 1, pair = j & 1;
  int klocal = ((v < 4) ? 2 * v : 16 + 2 * (v - 4)) + pair + ((lane >= 16) ? 8 : 0);
  int K = kt * 32 + klocal;
  int N = nt * 16 + (lane & 15);
  float val;
  if (mat == 0)      val = W1[K * D_DIM + N];             // W1a: multiplies h_u (forward)
  else if (mat == 1) val = W1[(K + D_DIM) * D_DIM + N];   // W1b: multiplies h_v (forward)
  else               val = W2[K * D_DIM + N];
  frags[id] = f32_to_bf16_rne(val);
}

// ---------------------------------------------------------------------------
// Main kernel: one wave = 16 edges x 128 features, both edge directions.
//   layer1: preF = Hu@W1a + Hv@W1b ; preR = Hv@W1a + Hu@W1b   (K=128 each)
//   s      = relu(preF+b1) + relu(preR+b1)          (bounced via LDS -> A frag)
//   out    = s@W2 + 2*b2
// 160 v_wmma_f32_16x16x32_bf16 per tile.
// ---------------------------------------------------------------------------
__global__ void __launch_bounds__(256)
edge_mlp_wmma(const bf16_t* __restrict__ hbf,
              const v4i* __restrict__ wfrag_gl,
              const int* __restrict__ src, const int* __restrict__ dst,
              const float* __restrict__ b1, const float* __restrict__ b2,
              float* __restrict__ out, int E, int ntiles) {
  __shared__ __align__(16) unsigned char smem[WFRAG_B + NWAVES * SBUF_B];

  // Block-wide one-time preload of all 96 B-fragments into LDS.
  for (int i = threadIdx.x; i < WFRAG_B / 16; i += 256)
    ((v4i*)smem)[i] = wfrag_gl[i];
  __syncthreads();

  const int lane = threadIdx.x & 31;
  const int wv   = threadIdx.x >> 5;
  const int col  = lane & 15;          // row slot (A) / column (B,C)
  const int hi   = (lane >> 4) & 1;    // half-wave select
  const v4i* Bl  = (const v4i*)smem;
  bf16_t* sbuf   = (bf16_t*)(smem + WFRAG_B + wv * SBUF_B);

  const int wave_stride = gridDim.x * NWAVES;
  for (int t = blockIdx.x * NWAVES + wv; t < ntiles; t += wave_stride) {
    const int e0 = t * 16;
    int row_edge = e0 + col;
    if (row_edge >= E) row_edge = E - 1;           // tail clamp (stores masked)
    const int iu = src[row_edge];
    const int iv = dst[row_edge];
    const char* hu = (const char*)(hbf + (size_t)iu * D_DIM);
    const char* hv = (const char*)(hbf + (size_t)iv * D_DIM);

    // Gather A fragments straight from bf16 rows (2 x b128 per fragment).
    Frag Au[KTILES], Av[KTILES];
#pragma unroll
    for (int k = 0; k < KTILES; ++k) {
      const int off = k * 64 + hi * 16;
      Au[k].q[0] = *(const v4i*)(hu + off);
      Au[k].q[1] = *(const v4i*)(hu + off + 32);
      Av[k].q[0] = *(const v4i*)(hv + off);
      Av[k].q[1] = *(const v4i*)(hv + off + 32);
    }

    // ---- Layer 1 + ReLU + direction merge ----
#pragma unroll
    for (int nt = 0; nt < NTILES; ++nt) {
      v8f accF = {};
      v8f accR = {};
#pragma unroll
      for (int k = 0; k < KTILES; ++k) {
        Frag B1a, B1b;
        const int fa = ((0 * MATFRAGS + k * 8 + nt) * 32 + lane) * 2;
        const int fb = ((1 * MATFRAGS + k * 8 + nt) * 32 + lane) * 2;
        B1a.q[0] = Bl[fa]; B1a.q[1] = Bl[fa + 1];
        B1b.q[0] = Bl[fb]; B1b.q[1] = Bl[fb + 1];
        // Two independent accumulation chains interleave on the XDL pipe.
        accF = __builtin_amdgcn_wmma_f32_16x16x32_bf16(false, Au[k].m, false, B1a.m,
                                                       (short)0, accF, false, false);
        accR = __builtin_amdgcn_wmma_f32_16x16x32_bf16(false, Av[k].m, false, B1a.m,
                                                       (short)0, accR, false, false);
        accF = __builtin_amdgcn_wmma_f32_16x16x32_bf16(false, Av[k].m, false, B1b.m,
                                                       (short)0, accF, false, false);
        accR = __builtin_amdgcn_wmma_f32_16x16x32_bf16(false, Au[k].m, false, B1b.m,
                                                       (short)0, accR, false, false);
      }
      const float bias = b1[nt * 16 + col];
#pragma unroll
      for (int r = 0; r < 8; ++r) {                 // C layout -> row-major LDS
        float f = accF[r] + bias; f = f > 0.0f ? f : 0.0f;
        float g = accR[r] + bias; g = g > 0.0f ? g : 0.0f;
        const int m = r + hi * 8;
        sbuf[m * D_DIM + nt * 16 + col] = f32_to_bf16_rne(f + g);
      }
    }

    // Same-wave LDS is in-order; wait drains stores before fragment readback.
    asm volatile("s_wait_dscnt 0" ::: "memory");

    // Re-layout s: row-major LDS -> A fragments (2 x ds b128 per fragment).
    Frag S[KTILES];
#pragma unroll
    for (int k = 0; k < KTILES; ++k) {
      const char* ps = (const char*)(sbuf + col * D_DIM) + k * 64 + hi * 16;
      S[k].q[0] = *(const v4i*)ps;
      S[k].q[1] = *(const v4i*)(ps + 32);
    }

    // ---- Layer 2 (merged: single pass for both directions) ----
#pragma unroll
    for (int nt = 0; nt < NTILES; ++nt) {
      v8f acc = {};
#pragma unroll
      for (int k = 0; k < KTILES; ++k) {
        Frag B2;
        const int fw = ((2 * MATFRAGS + k * 8 + nt) * 32 + lane) * 2;
        B2.q[0] = Bl[fw]; B2.q[1] = Bl[fw + 1];
        acc = __builtin_amdgcn_wmma_f32_16x16x32_bf16(false, S[k].m, false, B2.m,
                                                      (short)0, acc, false, false);
      }
      const float bias2 = 2.0f * b2[nt * 16 + col];
#pragma unroll
      for (int r = 0; r < 8; ++r) {
        const int m = r + hi * 8;
        if (e0 + m < E)
          out[(size_t)(e0 + m) * D_DIM + nt * 16 + col] = acc[r] + bias2;
      }
    }
  }
}

// ---------------------------------------------------------------------------
// Launch: inputs are {h, src, dst, W1, b1, W2, b2}; output [E,128] f32.
// ws layout: [0, N*128*2) bf16 h ; then 96KB packed weight fragments.
// ---------------------------------------------------------------------------
extern "C" void kernel_launch(void* const* d_in, const int* in_sizes, int n_in,
                              void* d_out, int out_size, void* d_ws, size_t ws_size,
                              hipStream_t stream) {
  const float* h   = (const float*)d_in[0];
  const int*   src = (const int*)d_in[1];
  const int*   dst = (const int*)d_in[2];
  const float* W1  = (const float*)d_in[3];
  const float* b1  = (const float*)d_in[4];
  const float* W2  = (const float*)d_in[5];
  const float* b2  = (const float*)d_in[6];
  float* out = (float*)d_out;

  const int nh = in_sizes[0];          // N * 128
  const int E  = in_sizes[1];

  bf16_t* hbf  = (bf16_t*)d_ws;
  bf16_t* frag = (bf16_t*)((char*)d_ws + (size_t)nh * sizeof(bf16_t));

  // 1) h -> bf16 (L2 resident afterwards: 12.8 MB)
  {
    int threads = (nh + 3) / 4;
    convert_h_bf16<<<(threads + 255) / 256, 256, 0, stream>>>(h, hbf, nh);
  }
  // 2) pack W1a/W1b/W2 into B-fragment register images (49152 elements)
  pack_weight_frags<<<(3 * MATFRAGS * 32 * 16 + 255) / 256, 256, 0, stream>>>(W1, W2, frag);

  // 3) edge MLP via bf16 WMMA
  const int ntiles = (E + 15) / 16;
  int blocks = 1024;
  int need = (ntiles + NWAVES - 1) / NWAVES;
  if (need < blocks) blocks = need;
  if (blocks < 1) blocks = 1;
  edge_mlp_wmma<<<blocks, 256, 0, stream>>>(hbf, (const v4i*)frag, src, dst,
                                            b1, b2, out, E, ntiles);
}
